// AttentionPooling_33706903339672
// MI455X (gfx1250) — matrix operand order repaired and checked by reading, not compiled, not run on hardware
//
#include <hip/hip_runtime.h>

typedef __attribute__((ext_vector_type(16))) __bf16 v16bf;
typedef __attribute__((ext_vector_type(8)))  float  v8f;
typedef unsigned int u32x4 __attribute__((ext_vector_type(4)));
typedef unsigned int u32x8 __attribute__((ext_vector_type(8)));

#define LSEQ 196
#define LPAD 224
#define PSTR 258   // bf16 row stride: odd dword count -> conflict-free column writes
#define USTR 260   // transposed-U row stride
#define CHUNK 128  // channels per TDM chunk

__device__ __forceinline__ unsigned short f2bf(float f) {
  unsigned u = __builtin_bit_cast(unsigned, f);
  unsigned r = u + 0x7FFFu + ((u >> 16) & 1u);
  return (unsigned short)(r >> 16);
}
__device__ __forceinline__ __bf16 bfbits(unsigned short h) {
  return __builtin_bit_cast(__bf16, h);
}

// TDM: load a 14 x 14 x tile_c fp32 tile (strides 1 / 224 / 50176 elements) to LDS.
// D# layout per CDNA5 ISA (group0/1/2/3 in SGPRs).
__device__ __forceinline__ void tdm_load_patch(const void* gaddr, unsigned lds_off,
                                               unsigned tile_c) {
  unsigned long long ga = (unsigned long long)(size_t)gaddr;
  u32x4 g0;
  g0[0] = 1u;                                     // count=1, user descriptor
  g0[1] = lds_off;                                // lds_addr
  g0[2] = (unsigned)ga;                           // global_addr[31:0]
  g0[3] = ((unsigned)(ga >> 32) & 0x01FFFFFFu)    // global_addr[56:32]
          | 0x80000000u;                          // type=2 ("image")
  u32x8 g1;
  g1[0] = 2u << 16;                               // data_size = 4 bytes
  g1[1] = 224u << 16;                             // tensor_dim0[15:0] (bits 63:48)
  g1[2] = 224u << 16;                             // tensor_dim1[15:0] (bits 95:80)
  g1[3] = 14u << 16;                              // tile_dim0 = 14   (bits 127:112)
  g1[4] = 14u | (tile_c << 16);                   // tile_dim1=14, tile_dim2
  g1[5] = 224u;                                   // tensor_dim0_stride[31:0]
  g1[6] = 50176u << 16;                           // tensor_dim1_stride[15:0] (bits 223:208)
  g1[7] = 0u;
  u32x4 g2; g2[0] = 256u; g2[1] = 0u; g2[2] = 0u; g2[3] = 0u;   // tensor_dim2
  u32x4 g3; g3[0] = 0u;  g3[1] = 0u; g3[2] = 0u; g3[3] = 0u;
  asm volatile("tensor_load_to_lds %0, %1, %2, %3"
               :: "s"(g0), "s"(g1), "s"(g2), "s"(g3)
               : "memory");
}

// ---------------- Kernel 1: fold query into transposed score matrix U^T[h][c] ----------------
// ws floats: [0..255] qh, [256..271] sbias (zero-padded), [272..] U^T ushort[16][260]
__global__ void prep_kernel(const float* __restrict__ query,
                            const float* __restrict__ in_w,
                            const float* __restrict__ in_b,
                            float* __restrict__ ws) {
  __shared__ float qh[256];
  const int t = threadIdx.x;
  float acc = in_b[t];
  for (int c = 0; c < 256; ++c) acc += query[c] * in_w[(size_t)t * 256 + c];
  qh[t] = acc;
  ws[t] = acc;
  __syncthreads();
  if (t < 16) {
    float s = 0.f;
    if (t < 8)
      for (int d = 0; d < 32; ++d) s += qh[t * 32 + d] * in_b[256 + t * 32 + d];
    ws[256 + t] = s;
  }
  unsigned short* UT = (unsigned short*)(ws + 272);
  for (int h = 0; h < 8; ++h) {
    float s = 0.f;
    for (int d = 0; d < 32; ++d)
      s += qh[h * 32 + d] * in_w[(size_t)(256 + h * 32 + d) * 256 + t];
    UT[h * USTR + t] = f2bf(s);
  }
  for (int h = 8; h < 16; ++h) UT[h * USTR + t] = 0;
  if (t < 64) UT[(t >> 2) * USTR + 256 + (t & 3)] = 0;   // pad columns
}

// ---------------- Kernel 2: one patch per block ----------------
__global__ void __launch_bounds__(256)
attnpool_kernel(const float* __restrict__ feat,
                const float* __restrict__ in_w,
                const float* __restrict__ in_b,
                const float* __restrict__ aow,
                const float* __restrict__ aob,
                const float* __restrict__ ow,
                const float* __restrict__ ob,
                const float* __restrict__ ws,
                float* __restrict__ out) {
  __shared__ unsigned short P[LPAD * PSTR];                 // 115584 B
  __shared__ unsigned short UT[16 * USTR];                  //   8320 B
  __shared__ float sb[16];
  __shared__ __align__(16) unsigned char scratch[CHUNK * LSEQ * 4];  // 100352 B
  // scratch reuse: Stage (Phase A) then Sc/Att/Pool/ctx/ao (Phases B..E)
  float*          Stage = (float*)scratch;
  float*          Sc    = (float*)scratch;                       // [16][224] f32
  unsigned short* Att   = (unsigned short*)(scratch + 14336);    // [16][224] bf16
  float*          Pool  = (float*)(scratch + 21504);             // [16][256] f32
  float*          ctxv  = (float*)(scratch + 37888);             // [256] f32
  float*          aov   = (float*)(scratch + 38912);             // [256] f32

  const int n  = blockIdx.x;
  const int bb = n >> 8;
  const int ph = (n >> 4) & 15;
  const int pw = n & 15;
  const int t    = threadIdx.x;
  const int wv   = t >> 5;
  const int lane = t & 31;
  const int half = lane >> 4;
  const int nn   = lane & 15;

  // ---- Phase A: TDM DMA patch -> LDS staging, convert f32 -> bf16 ----
  const float* fbase = feat + (size_t)bb * 256 * 50176
                            + (size_t)(ph * 14) * 224 + pw * 14;
  const unsigned stageOff = (unsigned)(size_t)&scratch[0];
  if (wv == 0) tdm_load_patch(fbase, stageOff, CHUNK);
  for (int i = t; i < (LPAD - LSEQ) * PSTR; i += 256) P[LSEQ * PSTR + i] = 0;
  for (int i = t; i < 16 * USTR; i += 256) UT[i] = ((const unsigned short*)(ws + 272))[i];
  if (t < 16) sb[t] = ws[256 + t];
  if (wv == 0) __builtin_amdgcn_s_wait_tensorcnt(0);
  __syncthreads();
  if (t < LSEQ) {
#pragma unroll 4
    for (int cc = 0; cc < CHUNK; cc += 2) {
      const unsigned lo = f2bf(Stage[cc * LSEQ + t]);
      const unsigned hi = f2bf(Stage[(cc + 1) * LSEQ + t]);
      *(unsigned*)&P[t * PSTR + cc] = lo | (hi << 16);
    }
  }
  __syncthreads();
  if (wv == 0) {
    tdm_load_patch(fbase + (size_t)CHUNK * 50176, stageOff, CHUNK);
    __builtin_amdgcn_s_wait_tensorcnt(0);
  }
  __syncthreads();
  if (t < LSEQ) {
#pragma unroll 4
    for (int cc = 0; cc < CHUNK; cc += 2) {
      const unsigned lo = f2bf(Stage[cc * LSEQ + t]);
      const unsigned hi = f2bf(Stage[(cc + 1) * LSEQ + t]);
      *(unsigned*)&P[t * PSTR + CHUNK + cc] = lo | (hi << 16);
    }
  }
  __syncthreads();

  // ---- Phase B: GEMM1 scores = P(208x256) x U(256x16), wmma bf16 ----
  const float scale = 0.17677669529663687f;     // 1/sqrt(32)
  for (int mt = wv; mt < 13; mt += 8) {
    v8f acc = {};
    const unsigned arow = (mt * 16 + nn) * PSTR;          // A: M = lane%16
    for (int kb = 0; kb < 8; ++kb) {
      const int k0 = kb * 32;
      const unsigned* ap = (const unsigned*)&P[arow + k0];
      const unsigned* bp = (const unsigned*)&UT[nn * USTR + k0];
      u32x8 au, bu;
#pragma unroll
      for (int v = 0; v < 8; ++v)                         // A K-halves per ISA layout
        au[v] = ap[(v >> 2) * 8 + half * 4 + (v & 3)];
#pragma unroll
      for (int j = 0; j < 8; ++j) bu[j] = bp[(j >> 2) * 8 + half * 4 + (j & 3)];  // B K-halves per ISA layout (U transposed)
      acc = __builtin_amdgcn_wmma_f32_16x16x32_bf16(
          false, __builtin_bit_cast(v16bf, au),
          false, __builtin_bit_cast(v16bf, bu),
          (short)0, acc, false, false);
    }
#pragma unroll
    for (int r = 0; r < 8; ++r)                           // D: M = half*8+r, N = lane%16
      Sc[nn * LPAD + mt * 16 + half * 8 + r] = acc[r] * scale + sb[nn];
  }
  __syncthreads();

  // ---- Phase C: softmax, one wave32 per head ----
  {
    float vals[7];
    float mx = -1e30f;
#pragma unroll
    for (int i = 0; i < 7; ++i) {
      const int l = lane + 32 * i;
      vals[i] = (l < LSEQ) ? Sc[wv * LPAD + l] : -1e30f;
      mx = fmaxf(mx, vals[i]);
    }
#pragma unroll
    for (int off = 16; off >= 1; off >>= 1) mx = fmaxf(mx, __shfl_xor(mx, off, 32));
    float s = 0.f;
#pragma unroll
    for (int i = 0; i < 7; ++i) {
      vals[i] = (lane + 32 * i < LSEQ) ? __expf(vals[i] - mx) : 0.f;
      s += vals[i];
    }
#pragma unroll
    for (int off = 16; off >= 1; off >>= 1) s += __shfl_xor(s, off, 32);
    const float inv = 1.0f / s;
#pragma unroll
    for (int i = 0; i < 7; ++i) {
      const int l = lane + 32 * i;
      Att[wv * LPAD + l] = (l < LSEQ) ? f2bf(vals[i] * inv) : (unsigned short)0;
    }
    for (int i = lane; i < LPAD; i += 32) Att[(wv + 8) * LPAD + i] = 0;  // pad heads
  }
  __syncthreads();

  // ---- Phase D: GEMM2 Pool(16x256) = Att(16x224) x P(224x256), wmma bf16 ----
  for (int ct = wv; ct < 16; ct += 8) {
    v8f acc = {};
    for (int kb = 0; kb < 7; ++kb) {
      const int l0 = kb * 32;
      const unsigned* ap = (const unsigned*)&Att[nn * LPAD + l0];
      u32x8 au;
#pragma unroll
      for (int v = 0; v < 8; ++v)
        au[v] = ap[(v >> 2) * 8 + half * 4 + (v & 3)];
      v16bf b;
#pragma unroll
      for (int e = 0; e < 16; ++e)                        // B: N = lane%16, K = (e/8)*16+half*8+e%8
        b[e] = bfbits(P[(l0 + (e >> 3) * 16 + half * 8 + (e & 7)) * PSTR + ct * 16 + nn]);
      acc = __builtin_amdgcn_wmma_f32_16x16x32_bf16(
          false, __builtin_bit_cast(v16bf, au),
          false, b, (short)0, acc, false, false);
    }
#pragma unroll
    for (int r = 0; r < 8; ++r)
      Pool[(half * 8 + r) * 256 + ct * 16 + nn] = acc[r];
  }
  __syncthreads();

  // ---- Phase E: ctx = Wv,h . Pool_h ; attn_out ; out (f32 FMA, L2-resident weights) ----
  {
    const int h = t >> 5;
    const float4* w4 = (const float4*)(in_w + (size_t)(512 + t) * 256);
    const float* pr = Pool + h * 256;
    float s = in_b[512 + t];
#pragma unroll 8
    for (int i = 0; i < 64; ++i) {
      const float4 w = w4[i];
      s += pr[4*i]*w.x + pr[4*i+1]*w.y + pr[4*i+2]*w.z + pr[4*i+3]*w.w;
    }
    ctxv[t] = s;
  }
  __syncthreads();
  {
    const float4* w4 = (const float4*)(aow + (size_t)t * 256);
    float s = aob[t];
#pragma unroll 8
    for (int i = 0; i < 64; ++i) {
      const float4 w = w4[i];
      s += ctxv[4*i]*w.x + ctxv[4*i+1]*w.y + ctxv[4*i+2]*w.z + ctxv[4*i+3]*w.w;
    }
    aov[t] = s;
  }
  __syncthreads();
  {
    const float4* w4 = (const float4*)(ow + (size_t)t * 256);
    float s = ob[t];
#pragma unroll 8
    for (int i = 0; i < 64; ++i) {
      const float4 w = w4[i];
      s += aov[4*i]*w.x + aov[4*i+1]*w.y + aov[4*i+2]*w.z + aov[4*i+3]*w.w;
    }
    out[(((size_t)bb * 256 + t) * 16 + ph) * 16 + pw] = s;
  }
}

extern "C" void kernel_launch(void* const* d_in, const int* in_sizes, int n_in,
                              void* d_out, int out_size, void* d_ws, size_t ws_size,
                              hipStream_t stream) {
  const float* feat  = (const float*)d_in[0];
  const float* query = (const float*)d_in[1];
  const float* in_w  = (const float*)d_in[2];
  const float* in_b  = (const float*)d_in[3];
  const float* aow   = (const float*)d_in[4];
  const float* aob   = (const float*)d_in[5];
  const float* ow    = (const float*)d_in[6];
  const float* ob    = (const float*)d_in[7];
  float* ws = (float*)d_ws;

  prep_kernel<<<1, 256, 0, stream>>>(query, in_w, in_b, ws);
  attnpool_kernel<<<2048, 256, 0, stream>>>(feat, in_w, in_b, aow, aob, ow, ob, ws,
                                            (float*)d_out);
}
